// MultiHeadDilatedAttention_48335561949404
// MI455X (gfx1250) — matrix operand order repaired and checked by reading, not compiled, not run on hardware
//
#include <hip/hip_runtime.h>
#include <math.h>

// ---------------- CDNA5 WMMA plumbing (gfx1250, wave32) ----------------
typedef __attribute__((ext_vector_type(16))) _Float16 v16h;
typedef __attribute__((ext_vector_type(8)))  float    v8f;

#define DMODEL 1024
#define DK     128
#define SEGC   128
#define TLEN   8192
#define BATCH  4
#define NORMF  0.08838834764831845f  // 1/sqrt(128)

// workspace layout (in _Float16 elements)
#define OFF_WQ 0u
#define OFF_WK 524288u
#define OFF_WV 1048576u
#define OFF_WO 1572864u
#define OFF_C  2097152u              // 32768 x 512 f16 concat buffer

// A-fragment (16x32 f16): lane = M row (ln%16); K groups of 8 interleaved by ln/16.
// caller passes p = row_base + k0 + (ln>>4)*8 ; halves at [0..7] and [16..23].
__device__ __forceinline__ v16h ldA(const _Float16* p) {
  union { v16h v; uint4 u[2]; } f;
  f.u[0] = *reinterpret_cast<const uint4*>(p);
  f.u[1] = *reinterpret_cast<const uint4*>(p + 16);
  return f.v;
}
// B-fragment (32x16 f16): lane = N col (ln%16); K groups of 8 interleaved by ln/16.
// caller passes p = col_base + k0 + (ln>>4)*8 ; halves at [0..7] and [16..23].
__device__ __forceinline__ v16h ldB(const _Float16* p) {
  union { v16h v; uint4 u[2]; } f;
  f.u[0] = *reinterpret_cast<const uint4*>(p);
  f.u[1] = *reinterpret_cast<const uint4*>(p + 16);
  return f.v;
}
__device__ __forceinline__ v8f wmma32(v16h a, v16h b, v8f c) {
  return __builtin_amdgcn_wmma_f32_16x16x32_f16(false, a, false, b, (short)0, c,
                                                false, false);
}

constexpr int smem_bytes_for(int DIL) {
  int L  = 64 / DIL;
  int Lp = (L < 16) ? 16 : L;
  int NP = (Lp < 32) ? 32 : Lp;
  int szXs = Lp * 72 * 2;
  int szSs = Lp * (Lp + 4) * 4;
  int szA  = ((szXs > szSs ? szXs : szSs) + 15) & ~15;
  int szQ  = (Lp * 136 * 2 + 15) & ~15;
  int szV  = 128 * (NP + 8) * 2;
  return szA + 2 * szQ + szV;
}

// ---------------- weight conversion: f32 -> f16 ----------------
__global__ void __launch_bounds__(256)
cvt_w(const float* __restrict__ Wk, const float* __restrict__ Wq,
      const float* __restrict__ Wv, const float* __restrict__ Wo,
      _Float16* __restrict__ ws) {
  const unsigned i = blockIdx.x * 256u + threadIdx.x;
  if (i < 524288u) {
    ws[OFF_WQ + i] = (_Float16)Wq[i];
    ws[OFF_WK + i] = (_Float16)Wk[i];
    ws[OFF_WV + i] = (_Float16)Wv[i];
    ws[OFF_WO + i] = (_Float16)Wo[i];
  }
}

// ---------------- per-head dilated attention ----------------
// One block per (b, s): computes Q/K/V, KQ^T, query-axis softmax, P^T V,
// and scatters into the f16 concat buffer at t = 64*s + DIL*l.
template <int DIL, int HEAD>
__global__ void __launch_bounds__(256)
attn_kernel(const float* __restrict__ x, const _Float16* __restrict__ wsh,
            _Float16* __restrict__ Cbuf) {
  constexpr int L  = 64 / DIL;
  constexpr int Lp = (L < 16) ? 16 : L;
  constexpr int MT = Lp / 16;
  constexpr int NP = (Lp < 32) ? 32 : Lp;        // padded K-dim for P^T * V
  constexpr int XS_STR = 72;                      // 64 + 8 halves (4-dword bank skew)
  constexpr int QK_STR = 136;                     // 128 + 8 halves
  constexpr int VT_STR = NP + 8;
  constexpr int S_STR  = Lp + 4;                  // dwords
  constexpr int PT_STR = NP + 8;
  constexpr int szXs = Lp * XS_STR * 2;
  constexpr int szSs = Lp * S_STR * 4;
  constexpr int szA  = ((szXs > szSs ? szXs : szSs) + 15) & ~15;
  constexpr int szQ  = (Lp * QK_STR * 2 + 15) & ~15;

  extern __shared__ char smem[];
  _Float16* Xs = reinterpret_cast<_Float16*>(smem);          // phase 1
  float*    Ss = reinterpret_cast<float*>(smem);             // phase 2/3 (overlays Xs)
  _Float16* Qs = reinterpret_cast<_Float16*>(smem + szA);    // phase 1/2
  _Float16* PT = Qs;                                         // phase 3/4 (overlays Qs)
  _Float16* Ks = reinterpret_cast<_Float16*>(smem + szA + szQ);
  _Float16* VT = reinterpret_cast<_Float16*>(smem + szA + 2 * szQ);

  const int tid = threadIdx.x;
  const int wv = tid >> 5;          // wave id 0..7
  const int ln = tid & 31;
  const int lr = ln & 15;
  const int hi = ln >> 4;

  const int s = blockIdx.x & 127;
  const int b = blockIdx.x >> 7;

  const float* xb = x + ((size_t)b * TLEN + s) * DMODEL;
  const _Float16* Wbase[3] = {
      wsh + OFF_WQ + (size_t)HEAD * DK * DMODEL,
      wsh + OFF_WK + (size_t)HEAD * DK * DMODEL,
      wsh + OFF_WV + (size_t)HEAD * DK * DMODEL};

  const v8f vzero = {0.f, 0.f, 0.f, 0.f, 0.f, 0.f, 0.f, 0.f};

  // ---- phase 1: Q/K/V = X_local * W^T, streamed over e in 64-wide chunks ----
  v8f acc[3 * MT];
#pragma unroll
  for (int i = 0; i < 3 * MT; ++i) acc[i] = vzero;

  for (int c = 0; c < DMODEL / 64; ++c) {
    __syncthreads();
    for (int idx = tid; idx < Lp * 64; idx += 256) {
      const int l = idx >> 6, e = idx & 63;
      float v = 0.f;
      if (l < L) v = xb[(size_t)l * DIL * SEGC * DMODEL + c * 64 + e];
      Xs[l * XS_STR + e] = (_Float16)v;
    }
    __syncthreads();
#pragma unroll
    for (int kk = 0; kk < 2; ++kk) {
      const int k0 = kk * 32;
      v16h af[MT];
#pragma unroll
      for (int mt = 0; mt < MT; ++mt)
        af[mt] = ldA(Xs + (mt * 16 + lr) * XS_STR + k0 + hi * 8);
#pragma unroll
      for (int wch = 0; wch < 3; ++wch) {
        const _Float16* wp =
            Wbase[wch] + (size_t)(wv * 16 + lr) * DMODEL + c * 64 + k0 + hi * 8;
        const v16h bf = ldB(wp);
#pragma unroll
        for (int mt = 0; mt < MT; ++mt)
          acc[wch * MT + mt] = wmma32(af[mt], bf, acc[wch * MT + mt]);
      }
    }
  }
  __syncthreads();
  // write-out: Q,K row-major [l][128]; V transposed [v][l] (contiguous 16B stores)
#pragma unroll
  for (int wch = 0; wch < 3; ++wch) {
#pragma unroll
    for (int mt = 0; mt < MT; ++mt) {
      const v8f a = acc[wch * MT + mt];
      if (wch < 2) {
        _Float16* dst = (wch == 0) ? Qs : Ks;
#pragma unroll
        for (int r = 0; r < 8; ++r)
          dst[(mt * 16 + hi * 8 + r) * QK_STR + wv * 16 + lr] = (_Float16)a[r];
      } else {
        alignas(16) _Float16 tmp[8];
#pragma unroll
        for (int r = 0; r < 8; ++r) tmp[r] = (_Float16)a[r];
        *reinterpret_cast<uint4*>(&VT[(wv * 16 + lr) * VT_STR + mt * 16 + hi * 8]) =
            *reinterpret_cast<const uint4*>(tmp);
      }
    }
  }
  __syncthreads();

  // ---- phase 2: S[n][m] = sum_d K[n][d] * Q[m][d] ----
  for (int t2 = wv; t2 < MT * MT; t2 += 8) {
    const int it = t2 / MT, jt = t2 % MT;
    v8f sa = vzero;
#pragma unroll
    for (int ks = 0; ks < 4; ++ks) {
      const v16h a  = ldA(Ks + (it * 16 + lr) * QK_STR + ks * 32 + hi * 8);
      const v16h bq = ldB(Qs + (jt * 16 + lr) * QK_STR + ks * 32 + hi * 8);
      sa = wmma32(a, bq, sa);
    }
#pragma unroll
    for (int r = 0; r < 8; ++r)
      Ss[(it * 16 + hi * 8 + r) * S_STR + jt * 16 + lr] = sa[r];
  }
  __syncthreads();

  // ---- phase 3: softmax along the QUERY axis (m) per key row n; P^T into LDS ----
  if (tid < NP) {
    const int n = tid;
    float mx = -3.4e38f, sm = 0.f;
    if (n < L) {
      for (int m = 0; m <= n; ++m) mx = fmaxf(mx, Ss[n * S_STR + m]);
      for (int m = 0; m <= n; ++m) sm += __expf((Ss[n * S_STR + m] - mx) * NORMF);
    }
    const float inv = (n < L) ? 1.f / sm : 0.f;
    for (int m = 0; m < Lp; ++m) {
      float p = 0.f;
      if (n < L && m <= n) p = __expf((Ss[n * S_STR + m] - mx) * NORMF) * inv;
      PT[m * PT_STR + n] = (_Float16)p;
    }
  }
  if (NP > Lp) {  // zero K-dim padding of V^T so padded WMMA reduction is exact
    for (int idx = tid; idx < 128 * (NP - Lp); idx += 256) {
      const int v = idx / (NP - Lp), j = idx % (NP - Lp);
      VT[v * VT_STR + Lp + j] = (_Float16)0.f;
    }
  }
  __syncthreads();

  // ---- phase 4: att[m][v] = sum_n P^T[m][n] * V[n][v]; scatter to concat buf ----
  for (int t4 = wv; t4 < MT * 8; t4 += 8) {
    const int mt = t4 >> 3;
    const int vt = wv;
    v8f aa = vzero;
#pragma unroll
    for (int ks = 0; ks < NP / 32; ++ks) {
      const v16h a  = ldA(PT + (mt * 16 + lr) * PT_STR + ks * 32 + hi * 8);
      const v16h bv = ldB(VT + (vt * 16 + lr) * VT_STR + ks * 32 + hi * 8);
      aa = wmma32(a, bv, aa);
    }
#pragma unroll
    for (int r = 0; r < 8; ++r) {
      const int m = mt * 16 + hi * 8 + r;
      if (m < L)
        Cbuf[((size_t)(b * TLEN + 64 * s + DIL * m)) * 512 + HEAD * 128 +
             vt * 16 + lr] = (_Float16)aa[r];
    }
  }
  if (DIL > 1) {  // 'same' padding: rows not hit by the strided scatter are zero
    for (int idx = tid; idx < 64 * 128; idx += 256) {
      const int toff = idx >> 7, v = idx & 127;
      if (toff % DIL)
        Cbuf[((size_t)(b * TLEN + 64 * s + toff)) * 512 + HEAD * 128 + v] =
            (_Float16)0.f;
    }
  }
}

// ---------------- output projection: (32768x512 f16) @ (512x1024) + bias ----------------
__global__ void __launch_bounds__(256)
proj_kernel(const _Float16* __restrict__ Cbuf, const _Float16* __restrict__ Woh,
            const float* __restrict__ bias, float* __restrict__ Out) {
  const int tid = threadIdx.x;
  const int wv = tid >> 5, ln = tid & 31, lr = ln & 15, hi = ln >> 4;
  const int m0 = blockIdx.x * 64 + (wv & 3) * 16;
  const int nbase = blockIdx.y * 128 + (wv >> 2) * 64;

  v8f acc[4];
#pragma unroll
  for (int j = 0; j < 4; ++j) {
    const float bv = bias[nbase + j * 16 + lr];
    const v8f t = {bv, bv, bv, bv, bv, bv, bv, bv};
    acc[j] = t;
  }
  const _Float16* arow = Cbuf + (size_t)(m0 + lr) * 512;
#pragma unroll 4
  for (int ks = 0; ks < 16; ++ks) {
    const v16h a = ldA(arow + ks * 32 + hi * 8);
#pragma unroll
    for (int j = 0; j < 4; ++j) {
      const v16h bf =
          ldB(Woh + (size_t)(nbase + j * 16 + lr) * 512 + ks * 32 + hi * 8);
      acc[j] = wmma32(a, bf, acc[j]);
    }
  }
#pragma unroll
  for (int j = 0; j < 4; ++j) {
    const int o = nbase + j * 16 + lr;
#pragma unroll
    for (int r = 0; r < 8; ++r)
      Out[(size_t)(m0 + hi * 8 + r) * DMODEL + o] = acc[j][r];
  }
}

// ---------------- launcher ----------------
extern "C" void kernel_launch(void* const* d_in, const int* in_sizes, int n_in,
                              void* d_out, int out_size, void* d_ws, size_t ws_size,
                              hipStream_t stream) {
  (void)in_sizes; (void)n_in; (void)out_size; (void)ws_size;
  const float* x  = (const float*)d_in[0];
  const float* Wk = (const float*)d_in[1];
  const float* Wq = (const float*)d_in[2];
  const float* Wv = (const float*)d_in[3];
  const float* Wo = (const float*)d_in[4];
  const float* bo = (const float*)d_in[5];
  float*      Out = (float*)d_out;
  _Float16*   wsh = (_Float16*)d_ws;

  // DIL=1 needs ~69KB dynamic LDS (<= 320KB WGP limit); raise the cap (idempotent).
  (void)hipFuncSetAttribute(reinterpret_cast<const void*>(&attn_kernel<1, 0>),
                            hipFuncAttributeMaxDynamicSharedMemorySize,
                            smem_bytes_for(1));
  (void)hipFuncSetAttribute(reinterpret_cast<const void*>(&attn_kernel<2, 1>),
                            hipFuncAttributeMaxDynamicSharedMemorySize,
                            smem_bytes_for(2));
  (void)hipFuncSetAttribute(reinterpret_cast<const void*>(&attn_kernel<4, 2>),
                            hipFuncAttributeMaxDynamicSharedMemorySize,
                            smem_bytes_for(4));
  (void)hipFuncSetAttribute(reinterpret_cast<const void*>(&attn_kernel<8, 3>),
                            hipFuncAttributeMaxDynamicSharedMemorySize,
                            smem_bytes_for(8));

  cvt_w<<<2048, 256, 0, stream>>>(Wk, Wq, Wv, Wo, wsh);

  const dim3 gA(BATCH * SEGC);
  attn_kernel<1, 0><<<gA, 256, smem_bytes_for(1), stream>>>(x, wsh, wsh + OFF_C);
  attn_kernel<2, 1><<<gA, 256, smem_bytes_for(2), stream>>>(x, wsh, wsh + OFF_C);
  attn_kernel<4, 2><<<gA, 256, smem_bytes_for(4), stream>>>(x, wsh, wsh + OFF_C);
  attn_kernel<8, 3><<<gA, 256, smem_bytes_for(8), stream>>>(x, wsh, wsh + OFF_C);

  const dim3 gP(512, 8);
  proj_kernel<<<gP, 256, 0, stream>>>(wsh + OFF_C, wsh + OFF_WO, bo, Out);
}
